// SubsetsSampleWeightedFormulaGRUHighway_71347996721716
// MI455X (gfx1250) — matrix operand order repaired and checked by reading, not compiled, not run on hardware
//
#include <hip/hip_runtime.h>
#include <hip/hip_bf16.h>
#include <stdint.h>

#define B_    16
#define A_    32
#define G_    512
#define S_    4096
#define TG_   1536
#define NE_   5
#define OH_   20
#define FE_   100
#define BINS_ 512
#define M_    8

typedef __bf16 bf16_t;
typedef __bf16 v16bf __attribute__((ext_vector_type(16)));
typedef float  v8f   __attribute__((ext_vector_type(8)));

__device__ inline v8f zero8() {
  v8f v = {0.f,0.f,0.f,0.f,0.f,0.f,0.f,0.f};
  return v;
}

// B-fragment (K=32 x N=16, bf16): lane n = lane&15, K-base = (lane>=16)*16,
// 16 contiguous K values per lane -> two 16B loads.
__device__ inline v16bf load_bfrag(const bf16_t* p) {
  union { v16bf v; uint4 q[2]; } u;
  u.q[0] = *(const uint4*)(p);
  u.q[1] = *(const uint4*)(p + 8);
  return u.v;
}

// A-fragment (M=16 x K=32, bf16) from a row-major row: lane m = lane&15,
// elems 0..15 = K kb..kb+15 (kb = k0 + hi*16).
__device__ inline v16bf load_afrag(const bf16_t* row, int kb) {
  union { v16bf v; uint4 q[2]; } u;
  u.q[0] = *(const uint4*)(row + kb);
  u.q[1] = *(const uint4*)(row + kb + 8);
  return u.v;
}

// ---------------- prep kernels ----------------

__global__ void k_cvt_bf16(const float* __restrict__ src, bf16_t* __restrict__ dst, int n) {
  int i = blockIdx.x * 256 + threadIdx.x;
  if (i < n) dst[i] = (bf16_t)src[i];
}

// mvfT[b][n][a] = bf16(vf[b,a,n] * mask[b,a])   (k-contiguous for B-frags)
__global__ void k_mvfT(const float* __restrict__ vf, const float* __restrict__ mask,
                       bf16_t* __restrict__ dst) {
  int i = blockIdx.x * 256 + threadIdx.x;
  if (i >= B_ * G_ * A_) return;
  int b = i / (G_ * A_);
  int r = i % (G_ * A_);
  int n = r / A_;
  int a = r % A_;
  dst[i] = (bf16_t)(vf[((size_t)b * A_ + a) * G_ + n] * mask[b * A_ + a]);
}

// CW[j][e*20+p] = sum_{q<=p} W_ih[j][e*20+q]
__global__ void k_cw(const float* __restrict__ Wih, float* __restrict__ CW) {
  int j = blockIdx.x * 256 + threadIdx.x;
  if (j >= TG_) return;
  const float* src = Wih + (size_t)j * FE_;
  float* dst = CW + (size_t)j * FE_;
  for (int e = 0; e < NE_; e++) {
    float run = 0.f;
    for (int p = 0; p < OH_; p++) {
      run += src[e * OH_ + p];
      dst[e * OH_ + p] = run;
    }
  }
}

// ---------------- K1: subset GEMM + counts + layernorm ----------------

__global__ void __launch_bounds__(256)
k_subset(const float* __restrict__ subs, const float* __restrict__ mask,
         const float* __restrict__ eloh, const bf16_t* __restrict__ mvfT,
         const float* __restrict__ lng, const float* __restrict__ lnb,
         bf16_t* __restrict__ normed, int* __restrict__ cnts) {
  int blk = blockIdx.x;
  int b = blk / (S_ / 16);
  int s0 = (blk % (S_ / 16)) * 16;
  int tid = threadIdx.x;
  int wave = tid >> 5, lane = tid & 31, m = lane & 15, hi = lane >> 4;

  __shared__ float smask[A_];
  __shared__ float sC[16][G_];
  __shared__ float sSize[16];
  __shared__ float sRedA[16][16];
  __shared__ float sRedB[16][16];
  __shared__ float sMean[16];
  __shared__ float sRstd[16];

  if (tid < A_) smask[tid] = mask[b * A_ + tid];
  __syncthreads();

  if (tid < 16) {
    int srow = b * S_ + s0 + tid;
    const float* sr = subs + (size_t)srow * A_;
    float size = 0.f;
    float c[NE_] = {0.f, 0.f, 0.f, 0.f, 0.f};
    for (int a = 0; a < A_; a++) {
      float sv = sr[a];
      size += sv * smask[a];
      const float* er = eloh + ((size_t)b * A_ + a) * NE_;
#pragma unroll
      for (int e = 0; e < NE_; e++) c[e] += er[e] * sv;
    }
    sSize[tid] = size + 1e-4f;
#pragma unroll
    for (int e = 0; e < NE_; e++) {
      int v = (int)c[e];
      v = v < 0 ? 0 : (v > OH_ - 1 ? OH_ - 1 : v);
      cnts[(size_t)srow * NE_ + e] = v;
    }
  }

  // A fragment: masked subset tile (16 rows x K=32 atoms), built from f32 global
  union { v16bf v; uint4 q[2]; } af;
  {
    const float* sr = subs + (size_t)(b * S_ + s0 + m) * A_;
    int kb = hi * 16;
#pragma unroll
    for (int j = 0; j < 8; j++) { int k = kb + j;      af.v[j]     = (bf16_t)(sr[k] * smask[k]); }
#pragma unroll
    for (int j = 0; j < 8; j++) { int k = kb + 8 + j; af.v[8 + j] = (bf16_t)(sr[k] * smask[k]); }
  }
  // 4 N-tiles per wave: swvs = subs_m x mvf   (single K=32 WMMA per tile)
#pragma unroll
  for (int t = 0; t < 4; t++) {
    int col = wave * 64 + t * 16 + m;
    v16bf bf = load_bfrag(mvfT + (size_t)b * G_ * A_ + (size_t)col * A_ + hi * 16);
    v8f c = zero8();
    c = __builtin_amdgcn_wmma_f32_16x16x32_bf16(false, af.v, false, bf, (short)0, c, false, false);
#pragma unroll
    for (int p = 0; p < 8; p++) sC[p + hi * 8][col] = c[p];
  }
  __syncthreads();

  // mean + layernorm (16 threads per row)
  int row = tid >> 4, lt = tid & 15;
  float inv_size = 1.f / sSize[row];
  float sum = 0.f, sq = 0.f;
  for (int j = lt; j < G_; j += 16) {
    float v = sC[row][j] * inv_size;
    sC[row][j] = v;
    sum += v; sq += v * v;
  }
  sRedA[row][lt] = sum; sRedB[row][lt] = sq;
  __syncthreads();
  if (lt == 0) {
    float ts = 0.f, tq = 0.f;
    for (int i = 0; i < 16; i++) { ts += sRedA[row][i]; tq += sRedB[row][i]; }
    float mean = ts * (1.f / G_);
    float var = tq * (1.f / G_) - mean * mean;
    sMean[row] = mean;
    sRstd[row] = rsqrtf(var + 1e-5f);
  }
  __syncthreads();
  float mean = sMean[row], rstd = sRstd[row];
  bf16_t* drow = normed + (size_t)(b * S_ + s0 + row) * G_;
  for (int j = lt; j < G_; j += 16) {
    float v = (sC[row][j] - mean) * rstd * lng[j] + lnb[j];
    drow[j] = (bf16_t)v;
  }
}

// ---------------- fused GRU + MLP + LN + score ----------------

__device__ inline void mlp_layer(const bf16_t (&src)[16][G_], const bf16_t* __restrict__ Wb,
                                 const float* __restrict__ bias, bf16_t (&dst)[16][G_],
                                 float (*fstage)[G_], int wave, int m, int hi) {
  v8f acc[4] = {zero8(), zero8(), zero8(), zero8()};
  for (int k0 = 0; k0 < G_; k0 += 32) {
    v16bf a = load_afrag(&src[m][0], k0 + hi * 16);
#pragma unroll
    for (int t = 0; t < 4; t++) {
      int n = wave * 64 + t * 16 + m;
      v16bf bf = load_bfrag(Wb + (size_t)n * G_ + k0 + hi * 16);
      acc[t] = __builtin_amdgcn_wmma_f32_16x16x32_bf16(false, a, false, bf, (short)0, acc[t], false, false);
    }
  }
#pragma unroll
  for (int t = 0; t < 4; t++) {
    int col = wave * 64 + t * 16 + m;
    float bv = bias[col];
#pragma unroll
    for (int p = 0; p < 8; p++) {
      int row = p + hi * 8;
      float v = acc[t][p] + bv;
      v = v > 0.f ? v : 0.f;
      dst[row][col] = (bf16_t)v;
      if (fstage) fstage[row][col] = v;
    }
  }
  __syncthreads();
}

__global__ void __launch_bounds__(256)
k_fused(const bf16_t* __restrict__ normed, const int* __restrict__ cnts,
        const float* __restrict__ CW, const float* __restrict__ b_ih,
        const float* __restrict__ b_hh, const bf16_t* __restrict__ Whh,
        const bf16_t* __restrict__ W1b, const bf16_t* __restrict__ W2ab,
        const bf16_t* __restrict__ W2bb, const float* __restrict__ b1,
        const float* __restrict__ b2a, const float* __restrict__ b2b,
        const float* __restrict__ lpg, const float* __restrict__ lpb,
        const float* __restrict__ Ws, const float* __restrict__ bs,
        float* __restrict__ scores) {
  int blk = blockIdx.x;
  int b = blk / (S_ / 16);
  int s0 = (blk % (S_ / 16)) * 16;
  int tid = threadIdx.x;
  int wave = tid >> 5, lane = tid & 31, m = lane & 15, hi = lane >> 4;

  __shared__ bf16_t sA[16][G_];
  __shared__ bf16_t sB[16][G_];
  __shared__ float  sF[16][G_];
  __shared__ int    sV[16][8];
  __shared__ float  sRedA[16][16];
  __shared__ float  sRedB[16][16];
  __shared__ float  sMean[16];
  __shared__ float  sRstd[16];

  // load normed tile (16 x 512 bf16, rows contiguous)
  {
    const uint4* src = (const uint4*)(normed + (size_t)(b * S_ + s0) * G_);
    uint4* dst = (uint4*)(&sA[0][0]);
    for (int i = tid; i < 16 * G_ / 8; i += 256) dst[i] = src[i];
  }
  if (tid < 16 * NE_) sV[tid / NE_][tid % NE_] = cnts[(size_t)(b * S_ + s0 + tid / NE_) * NE_ + tid % NE_];
  __syncthreads();

  // gh = h @ W_hh^T   (each wave: cols [wave*64, wave*64+64) of gate space, 3 gates)
  v8f accR[4] = {zero8(), zero8(), zero8(), zero8()};
  v8f accZ[4] = {zero8(), zero8(), zero8(), zero8()};
  v8f accN[4] = {zero8(), zero8(), zero8(), zero8()};
  for (int k0 = 0; k0 < G_; k0 += 32) {
    v16bf a = load_afrag(&sA[m][0], k0 + hi * 16);
#pragma unroll
    for (int t = 0; t < 4; t++) {
      int n = wave * 64 + t * 16 + m;
      v16bf br = load_bfrag(Whh + (size_t)n * G_ + k0 + hi * 16);
      accR[t] = __builtin_amdgcn_wmma_f32_16x16x32_bf16(false, a, false, br, (short)0, accR[t], false, false);
      v16bf bz = load_bfrag(Whh + (size_t)(n + 512) * G_ + k0 + hi * 16);
      accZ[t] = __builtin_amdgcn_wmma_f32_16x16x32_bf16(false, a, false, bz, (short)0, accZ[t], false, false);
      v16bf bn = load_bfrag(Whh + (size_t)(n + 1024) * G_ + k0 + hi * 16);
      accN[t] = __builtin_amdgcn_wmma_f32_16x16x32_bf16(false, a, false, bn, (short)0, accN[t], false, false);
    }
  }

  // GRU combine: gi from prefix-summed W_ih gathers (thermometer one-hot GEMM)
#pragma unroll
  for (int t = 0; t < 4; t++) {
    int col = wave * 64 + t * 16 + m;
    float bir = b_ih[col], biz = b_ih[col + 512], bin = b_ih[col + 1024];
    float bhr = b_hh[col], bhz = b_hh[col + 512], bhn = b_hh[col + 1024];
#pragma unroll
    for (int p = 0; p < 8; p++) {
      int row = p + hi * 8;
      float gir = bir, giz = biz, gin = bin;
#pragma unroll
      for (int e = 0; e < NE_; e++) {
        int idx = e * OH_ + sV[row][e];
        gir += CW[(size_t)col * FE_ + idx];
        giz += CW[(size_t)(col + 512) * FE_ + idx];
        gin += CW[(size_t)(col + 1024) * FE_ + idx];
      }
      float r = 1.f / (1.f + __expf(-(gir + (accR[t][p] + bhr))));
      float z = 1.f / (1.f + __expf(-(giz + (accZ[t][p] + bhz))));
      float nn = tanhf(gin + r * (accN[t][p] + bhn));
      float hold = (float)sA[row][col];
      sB[row][col] = (bf16_t)((1.f - z) * nn + z * hold);
    }
  }
  __syncthreads();

  // 3-layer ReLU MLP (ping-pong LDS buffers), last layer also staged in f32
  mlp_layer(sB, W1b,  b1,  sA, (float(*)[G_])nullptr, wave, m, hi);
  mlp_layer(sA, W2ab, b2a, sB, (float(*)[G_])nullptr, wave, m, hi);
  mlp_layer(sB, W2bb, b2b, sA, sF, wave, m, hi);

  // layernorm + score (16 threads per row)
  int row = tid >> 4, lt = tid & 15;
  float sum = 0.f, sq = 0.f;
  for (int j = lt; j < G_; j += 16) { float v = sF[row][j]; sum += v; sq += v * v; }
  sRedA[row][lt] = sum; sRedB[row][lt] = sq;
  __syncthreads();
  if (lt == 0) {
    float ts = 0.f, tq = 0.f;
    for (int i = 0; i < 16; i++) { ts += sRedA[row][i]; tq += sRedB[row][i]; }
    float mean = ts * (1.f / G_);
    float var = tq * (1.f / G_) - mean * mean;
    sMean[row] = mean;
    sRstd[row] = rsqrtf(var + 1e-5f);
  }
  __syncthreads();
  float mean = sMean[row], rstd = sRstd[row];
  float sc = 0.f;
  for (int j = lt; j < G_; j += 16) {
    float v = (sF[row][j] - mean) * rstd * lpg[j] + lpb[j];
    sc += v * Ws[j];
  }
  sRedA[row][lt] = sc;
  __syncthreads();
  if (lt == 0) {
    float tot = 0.f;
    for (int i = 0; i < 16; i++) tot += sRedA[row][i];
    scores[(size_t)b * S_ + s0 + row] = tot + bs[0];
  }
}

// ---------------- softmax + histogram ----------------

__global__ void k_softmax(const float* __restrict__ scores, float* __restrict__ probs) {
  int b = blockIdx.x, tid = threadIdx.x;
  __shared__ float red[256];
  float mx = -3.4e38f;
  for (int s = tid; s < S_; s += 256) mx = fmaxf(mx, scores[(size_t)b * S_ + s]);
  red[tid] = mx; __syncthreads();
  for (int o = 128; o > 0; o >>= 1) { if (tid < o) red[tid] = fmaxf(red[tid], red[tid + o]); __syncthreads(); }
  mx = red[0]; __syncthreads();
  float sum = 0.f;
  for (int s = tid; s < S_; s += 256) sum += __expf(scores[(size_t)b * S_ + s] - mx);
  red[tid] = sum; __syncthreads();
  for (int o = 128; o > 0; o >>= 1) { if (tid < o) red[tid] += red[tid + o]; __syncthreads(); }
  float inv = 1.f / red[0];
  for (int s = tid; s < S_; s += 256)
    probs[(size_t)b * S_ + s] = __expf(scores[(size_t)b * S_ + s] - mx) * inv;
}

__global__ void k_hist(const float* __restrict__ peaks, const float* __restrict__ probs,
                       float* __restrict__ spect) {
  int b = blockIdx.x, tid = threadIdx.x;
  __shared__ float hb[BINS_];
  for (int j = tid; j < BINS_; j += 256) hb[j] = 0.f;
  __syncthreads();
  for (int idx = tid; idx < S_ * M_; idx += 256) {
    int s = idx / M_, mm = idx % M_;
    const float* pk = peaks + (((size_t)b * S_ + s) * M_ + mm) * 2;
    int bin = (int)rintf(pk[0]);  // round-half-even, matches jnp.round
    bin = bin < 0 ? 0 : (bin > BINS_ - 1 ? BINS_ - 1 : bin);
    atomicAdd(&hb[bin], pk[1] * probs[(size_t)b * S_ + s]);
  }
  __syncthreads();
  for (int j = tid; j < BINS_; j += 256) spect[(size_t)b * BINS_ + j] = hb[j];
}

// ---------------- launch ----------------

extern "C" void kernel_launch(void* const* d_in, const int* in_sizes, int n_in,
                              void* d_out, int out_size, void* d_ws, size_t ws_size,
                              hipStream_t stream) {
  const float* vf    = (const float*)d_in[0];
  const float* mask  = (const float*)d_in[1];
  const float* eloh  = (const float*)d_in[2];
  const float* subs  = (const float*)d_in[4];
  const float* peaks = (const float*)d_in[5];
  const float* lng   = (const float*)d_in[6];
  const float* lnb   = (const float*)d_in[7];
  const float* Wih   = (const float*)d_in[8];
  const float* Whh_f = (const float*)d_in[9];
  const float* b_ih  = (const float*)d_in[10];
  const float* b_hh  = (const float*)d_in[11];
  const float* W1_f  = (const float*)d_in[12];
  const float* b1    = (const float*)d_in[13];
  const float* W2a_f = (const float*)d_in[14];
  const float* b2a   = (const float*)d_in[15];
  const float* W2b_f = (const float*)d_in[16];
  const float* b2b   = (const float*)d_in[17];
  const float* lpg   = (const float*)d_in[18];
  const float* lpb   = (const float*)d_in[19];
  const float* Ws    = (const float*)d_in[20];
  const float* bs    = (const float*)d_in[21];

  char* ws = (char*)d_ws;
  size_t off = 0;
  bf16_t* mvfT   = (bf16_t*)(ws + off); off += (size_t)B_ * G_ * A_ * 2;       // 0.5 MB
  bf16_t* Whh_b  = (bf16_t*)(ws + off); off += (size_t)TG_ * G_ * 2;           // 1.5 MB
  bf16_t* W1_b   = (bf16_t*)(ws + off); off += (size_t)G_ * G_ * 2;
  bf16_t* W2a_b  = (bf16_t*)(ws + off); off += (size_t)G_ * G_ * 2;
  bf16_t* W2b_b  = (bf16_t*)(ws + off); off += (size_t)G_ * G_ * 2;
  float*  CW     = (float*)(ws + off);  off += (size_t)TG_ * FE_ * 4;          // 0.6 MB
  int*    cnts   = (int*)(ws + off);    off += (size_t)B_ * S_ * NE_ * 4;      // 1.3 MB
  float*  scores = (float*)(ws + off);  off += (size_t)B_ * S_ * 4;            // 0.25 MB
  bf16_t* normed = (bf16_t*)(ws + off); off += (size_t)B_ * S_ * G_ * 2;       // 64 MB

  float* spect = (float*)d_out;                  // [B, 512]
  float* probs = (float*)d_out + B_ * BINS_;     // [B, S]

  // prep: weight conversions + prefix sums
  k_cvt_bf16<<<(TG_ * G_ + 255) / 256, 256, 0, stream>>>(Whh_f, Whh_b, TG_ * G_);
  k_cvt_bf16<<<(G_ * G_ + 255) / 256, 256, 0, stream>>>(W1_f,  W1_b,  G_ * G_);
  k_cvt_bf16<<<(G_ * G_ + 255) / 256, 256, 0, stream>>>(W2a_f, W2a_b, G_ * G_);
  k_cvt_bf16<<<(G_ * G_ + 255) / 256, 256, 0, stream>>>(W2b_f, W2b_b, G_ * G_);
  k_mvfT<<<(B_ * G_ * A_ + 255) / 256, 256, 0, stream>>>(vf, mask, mvfT);
  k_cw<<<(TG_ + 255) / 256, 256, 0, stream>>>(Wih, CW);

  // main pipeline
  int nblk = B_ * (S_ / 16);
  k_subset<<<nblk, 256, 0, stream>>>(subs, mask, eloh, mvfT, lng, lnb, normed, cnts);
  k_fused<<<nblk, 256, 0, stream>>>(normed, cnts, CW, b_ih, b_hh, Whh_b,
                                    W1_b, W2a_b, W2b_b, b1, b2a, b2b,
                                    lpg, lpb, Ws, bs, scores);
  k_softmax<<<B_, 256, 0, stream>>>(scores, probs);
  k_hist<<<B_, 256, 0, stream>>>(peaks, probs, spect);
}
